// GCN_50053548867611
// MI455X (gfx1250) — matrix operand order repaired and checked by reading, not compiled, not run on hardware
//
#include <hip/hip_runtime.h>
#include <stdint.h>

// ---------------------------------------------------------------------------
// Types for CDNA5 WMMA (wave32): 16x16x32 bf16 -> f32
// ---------------------------------------------------------------------------
typedef __attribute__((ext_vector_type(16))) __bf16 v16bf;
typedef __attribute__((ext_vector_type(8)))  float  v8f;
typedef __attribute__((ext_vector_type(4)))  int    v4i;

#define NNODES 20000
#define MPAD   20096      // 157 * 128 : padded row count, no bounds checks in GEMM
#define NEDGES 320000
#define NGRAPH 64
#define HDIM   256
#define LDSW   36         // LDS row stride in dwords (144 B: 16B-aligned, bank-friendly)

// Async global->LDS copy support (gfx1250 ASYNCcnt path), compile-probed.
#if defined(__has_builtin)
#  if __has_builtin(__builtin_amdgcn_global_load_async_to_lds_b128)
#    define ASYNC_LDS 1
#  endif
#endif
#ifndef ASYNC_LDS
#  define ASYNC_LDS 0
#endif

typedef __attribute__((address_space(1))) v4i gv4i;   // global int4
typedef __attribute__((address_space(3))) v4i sv4i;   // LDS int4

__device__ __forceinline__ void atomAddF(float* p, float v) {
    __hip_atomic_fetch_add(p, v, __ATOMIC_RELAXED, __HIP_MEMORY_SCOPE_AGENT);
}

// 16-byte global -> LDS copy (async when available)
__device__ __forceinline__ void cp16(const uint32_t* g, uint32_t* l) {
#if ASYNC_LDS
    __builtin_amdgcn_global_load_async_to_lds_b128(
        (gv4i*)g, (sv4i*)l, 0, 0);
#else
    *(uint4*)l = *(const uint4*)g;
#endif
}

// wait on outstanding async copies: allow `more` in flight (3 = next tile's ops)
__device__ __forceinline__ void waitAsync(bool more) {
#if ASYNC_LDS
#  if __has_builtin(__builtin_amdgcn_s_wait_asynccnt)
    if (more) __builtin_amdgcn_s_wait_asynccnt(3);
    else      __builtin_amdgcn_s_wait_asynccnt(0);
#  else
    if (more) asm volatile("s_wait_asynccnt 0x3" ::: "memory");
    else      asm volatile("s_wait_asynccnt 0x0" ::: "memory");
#  endif
#endif
}

// ---------------------------------------------------------------------------
// Degree / normalization
// ---------------------------------------------------------------------------
__global__ __launch_bounds__(256) void degree_k(const int* __restrict__ dst,
                                                float* __restrict__ deg, int E) {
    int e = blockIdx.x * 256 + threadIdx.x;
    if (e < E) atomAddF(&deg[dst[e]], 1.0f);
}

__global__ __launch_bounds__(256) void dinv_k(const float* __restrict__ deg,
                                              float* __restrict__ dinv, int n) {
    int i = blockIdx.x * 256 + threadIdx.x;
    if (i < n) dinv[i] = rsqrtf(deg[i] + 1.0f);  // +1 self loop; always > 0
}

// ---------------------------------------------------------------------------
// fp32 -> bf16 conversion kernels (write MPAD rows; pad rows are zero)
// ---------------------------------------------------------------------------
__global__ __launch_bounds__(256) void convert_x(const float* __restrict__ x,
                                                 __bf16* __restrict__ hb) {
    int idx = blockIdx.x * 256 + threadIdx.x;
    if (idx >= MPAD * 32) return;
    int n = idx >> 5, k = idx & 31;
    float v = (n < NNODES && k < 24) ? x[n * 24 + k] : 0.0f;
    hb[idx] = (__bf16)v;
}

__global__ __launch_bounds__(256) void convert_h(const float* __restrict__ agg,
                                                 const float* __restrict__ bias,
                                                 __bf16* __restrict__ hb) {
    int idx = blockIdx.x * 256 + threadIdx.x;
    if (idx >= MPAD * HDIM) return;
    int n = idx >> 8;
    float v = (n < NNODES) ? fmaxf(agg[idx] + bias[idx & 255], 0.0f) : 0.0f;
    hb[idx] = (__bf16)v;
}

// W (K x 256) fp32 -> BT (256 x Kp) bf16 TRANSPOSED, zero-pad cols K..Kp-1
__global__ __launch_bounds__(256) void convert_w(const float* __restrict__ W,
                                                 __bf16* __restrict__ BT, int K, int Kp) {
    int idx = blockIdx.x * 256 + threadIdx.x;
    if (idx >= 256 * Kp) return;
    int n = idx / Kp, k = idx - n * Kp;
    float v = (k < K) ? W[k * 256 + n] : 0.0f;
    BT[idx] = (__bf16)v;
}

// ---------------------------------------------------------------------------
// WMMA GEMM: C[MPAD x 256] f32 = A[MPAD x K] bf16 @ BT[256 x K]^T bf16
// Block: 256 threads = 8 waves. Block tile 128x64, wave tile 32x32.
// Double-buffered LDS; async global->LDS staging overlaps WMMA compute.
// ---------------------------------------------------------------------------
__global__ __launch_bounds__(256) void gemm_bf16(const __bf16* __restrict__ A,
                                                 const __bf16* __restrict__ BT,
                                                 float* __restrict__ C,
                                                 int K) {
    __shared__ uint32_t As32[2][128][LDSW];   // 2 x (128 x 32 bf16), stride 72 bf16
    __shared__ uint32_t Bs32[2][64][LDSW];    // 2 x (64 n-rows x 32 bf16)

    const int tid  = threadIdx.x;
    const int wave = tid >> 5;
    const int lane = tid & 31;
    const int wm   = wave >> 1;            // 0..3 : 32-row group
    const int wn   = wave & 1;             // 0..1 : 32-col group
    const int m0   = blockIdx.y * 128;
    const int n0   = blockIdx.x * 64;

    v8f acc[2][2] = {};

    const int ld32 = K >> 1;               // global row stride in dwords
    const int arow = tid >> 1;             // A staging: row 0..127, dword col 0/8
    const int acol = (tid & 1) * 8;
    const int brow = tid >> 2;             // B staging: n-row 0..63, dword col 0/4/8/12
    const int bcol = (tid & 3) * 4;

    const uint32_t* Ag = (const uint32_t*)A + (size_t)(m0 + arow) * ld32 + acol;
    const uint32_t* Bg = (const uint32_t*)BT + (size_t)(n0 + brow) * ld32 + bcol;

    const int mlsel = lane & 15;
    const int hsel  = lane >> 4;

    const int ntiles = K >> 5;

    // prologue: stage tile 0 into buffer 0
    cp16(Ag,     &As32[0][arow][acol]);
    cp16(Ag + 4, &As32[0][arow][acol + 4]);
    cp16(Bg,     &Bs32[0][brow][bcol]);

    for (int it = 0; it < ntiles; ++it) {
        const int buf = it & 1;
        const bool more = (it + 1) < ntiles;
        if (more) {                         // stage next tile into other buffer
            const int kd = (it + 1) << 4;   // dword offset along K
            cp16(Ag + kd,     &As32[buf ^ 1][arow][acol]);
            cp16(Ag + kd + 4, &As32[buf ^ 1][arow][acol + 4]);
            cp16(Bg + kd,     &Bs32[buf ^ 1][brow][bcol]);
        }
        waitAsync(more);                    // current tile's copies complete
        __syncthreads();                    // visible to all waves

        const __bf16* Asb = (const __bf16*)As32[buf];
        const __bf16* Bsb = (const __bf16*)Bs32[buf];

        // --- build fragments per ISA 16-bit layouts
        v16bf af[2], bfr[2];
#pragma unroll
        for (int i = 0; i < 2; ++i) {
            const int mrow = wm * 32 + i * 16 + mlsel;
#pragma unroll
            for (int v = 0; v < 8; ++v) {
                const int kk = ((v < 4) ? 0 : 16) + hsel * 8 + (v & 3) * 2;
                af[i][2 * v]     = Asb[mrow * (2 * LDSW) + kk];
                af[i][2 * v + 1] = Asb[mrow * (2 * LDSW) + kk + 1];
            }
        }
#pragma unroll
        for (int jj = 0; jj < 2; ++jj) {
            const int nrow = wn * 32 + jj * 16 + mlsel;
#pragma unroll
            for (int v = 0; v < 8; ++v) {
                const int kk = ((v < 4) ? 0 : 16) + hsel * 8 + (v & 3) * 2;
                bfr[jj][2 * v]     = Bsb[nrow * (2 * LDSW) + kk];
                bfr[jj][2 * v + 1] = Bsb[nrow * (2 * LDSW) + kk + 1];
            }
        }

#pragma unroll
        for (int i = 0; i < 2; ++i)
#pragma unroll
            for (int jj = 0; jj < 2; ++jj)
                acc[i][jj] = __builtin_amdgcn_wmma_f32_16x16x32_bf16(
                    false, af[i], false, bfr[jj], (short)0, acc[i][jj], false, false);

        __syncthreads();                    // reads done: buffer safe to overwrite
    }

    // --- store C (f32 16x16 C/D layout), unconditional (C padded to MPAD)
    const int mb = hsel * 8;
#pragma unroll
    for (int i = 0; i < 2; ++i) {
#pragma unroll
        for (int jj = 0; jj < 2; ++jj) {
            const int col = n0 + wn * 32 + jj * 16 + mlsel;
            float* cp = C + (size_t)(m0 + wm * 32 + i * 16 + mb) * 256 + col;
#pragma unroll
            for (int r = 0; r < 8; ++r)
                cp[(size_t)r * 256] = acc[i][jj][r];
        }
    }
}

// ---------------------------------------------------------------------------
// Edge aggregation: one wave per message (E edges + N self loops).
// agg[dst] += norm * t[src]   (fp32 hardware atomics)
// ---------------------------------------------------------------------------
__global__ __launch_bounds__(256) void aggregate_k(const float* __restrict__ t,
                                                   float* __restrict__ agg,
                                                   const int* __restrict__ src,
                                                   const int* __restrict__ dst,
                                                   const float* __restrict__ dinv,
                                                   int E, int n) {
    const int wave = blockIdx.x * 8 + (threadIdx.x >> 5);
    const int lane = threadIdx.x & 31;
    if (wave >= E + n) return;

    int s, d; float nrm;
    if (wave < E) {
        s = src[wave]; d = dst[wave];
        nrm = dinv[s] * dinv[d];
    } else {
        s = d = wave - E;
        float x = dinv[s];
        nrm = x * x;
    }

    const float4* tp = (const float4*)(t + (size_t)s * HDIM) + lane * 2;
    float4 a = tp[0];
    float4 b = tp[1];
    float* out = agg + (size_t)d * HDIM + lane * 8;
    atomAddF(out + 0, a.x * nrm);
    atomAddF(out + 1, a.y * nrm);
    atomAddF(out + 2, a.z * nrm);
    atomAddF(out + 3, a.w * nrm);
    atomAddF(out + 4, b.x * nrm);
    atomAddF(out + 5, b.y * nrm);
    atomAddF(out + 6, b.z * nrm);
    atomAddF(out + 7, b.w * nrm);
}

// ---------------------------------------------------------------------------
// Global mean pool (sum + count), bias b6 folded in at the head
// ---------------------------------------------------------------------------
__global__ __launch_bounds__(256) void pool_k(const float* __restrict__ agg,
                                              const int* __restrict__ batch,
                                              float* __restrict__ psum,
                                              float* __restrict__ cnt) {
    const int nnode = blockIdx.x;
    const int c = threadIdx.x;
    const int g = batch[nnode];
    atomAddF(&psum[g * HDIM + c], agg[(size_t)nnode * HDIM + c]);
    if (c == 0) atomAddF(&cnt[g], 1.0f);
}

// ---------------------------------------------------------------------------
// Head: pooled = (psum + cnt*b6)/max(cnt,1); logits = pooled@Wlin + blin;
// log_softmax over 2 classes. One block per graph.
// ---------------------------------------------------------------------------
__global__ __launch_bounds__(256) void head_k(const float* __restrict__ psum,
                                              const float* __restrict__ cnt,
                                              const float* __restrict__ b6,
                                              const float* __restrict__ Wlin,
                                              const float* __restrict__ blin,
                                              float* __restrict__ out) {
    __shared__ float red0[256];
    __shared__ float red1[256];
    const int g = blockIdx.x;
    const int c = threadIdx.x;
    const float cn = cnt[g];
    const float denom = fmaxf(cn, 1.0f);
    const float val = (psum[g * HDIM + c] + cn * b6[c]) / denom;
    red0[c] = val * Wlin[c * 2 + 0];
    red1[c] = val * Wlin[c * 2 + 1];
    __syncthreads();
    for (int s = 128; s > 0; s >>= 1) {
        if (c < s) { red0[c] += red0[c + s]; red1[c] += red1[c + s]; }
        __syncthreads();
    }
    if (c == 0) {
        float l0 = red0[0] + blin[0];
        float l1 = red1[0] + blin[1];
        float mx = fmaxf(l0, l1);
        float lse = mx + logf(expf(l0 - mx) + expf(l1 - mx));
        out[g * 2 + 0] = l0 - lse;
        out[g * 2 + 1] = l1 - lse;
    }
}

// ---------------------------------------------------------------------------
// Launch
// ---------------------------------------------------------------------------
extern "C" void kernel_launch(void* const* d_in, const int* in_sizes, int n_in,
                              void* d_out, int out_size, void* d_ws, size_t ws_size,
                              hipStream_t stream) {
    const int N = NNODES, E = NEDGES, G = NGRAPH;

    const float* x     = (const float*)d_in[0];
    const int*   edge  = (const int*)d_in[1];     // [2,E]: row0=src, row1=dst
    const int*   batch = (const int*)d_in[2];
    const float* Ws[6] = {(const float*)d_in[3], (const float*)d_in[5],
                          (const float*)d_in[7], (const float*)d_in[9],
                          (const float*)d_in[11], (const float*)d_in[13]};
    const float* bs[6] = {(const float*)d_in[4], (const float*)d_in[6],
                          (const float*)d_in[8], (const float*)d_in[10],
                          (const float*)d_in[12], (const float*)d_in[14]};
    const float* Wlin = (const float*)d_in[15];
    const float* blin = (const float*)d_in[16];
    const int* srcp = edge;
    const int* dstp = edge + E;

    // carve workspace (256-byte aligned chunks)
    size_t off = 0;
    char* base = (char*)d_ws;
    auto carve = [&](size_t bytes) -> void* {
        void* p = base + off;
        off += (bytes + 255) & ~(size_t)255;
        return p;
    };
    __bf16* hb   = (__bf16*)carve((size_t)MPAD * HDIM * 2);
    __bf16* wb   = (__bf16*)carve((size_t)HDIM * HDIM * 2);   // transposed weights
    float*  tbuf = (float*)carve((size_t)MPAD * HDIM * 4);
    float*  agg  = (float*)carve((size_t)N * HDIM * 4);
    float*  deg  = (float*)carve((size_t)N * 4);
    float*  dinv = (float*)carve((size_t)N * 4);
    float*  psum = (float*)carve((size_t)G * HDIM * 4);
    float*  cnt  = (float*)carve((size_t)G * 4);
    (void)ws_size; (void)n_in; (void)in_sizes; (void)out_size;

    const dim3 gemmGrid(HDIM / 64, MPAD / 128);   // (4, 157)
    const int aggBlocks = (E + N + 7) / 8;

    // degrees + symmetric norm factors
    (void)hipMemsetAsync(deg, 0, (size_t)N * 4, stream);
    degree_k<<<(E + 255) / 256, 256, 0, stream>>>(dstp, deg, E);
    dinv_k<<<(N + 255) / 256, 256, 0, stream>>>(deg, dinv, N);

    // ----- layer 1: x (N x 24) -> agg (N x 256)
    convert_x<<<(MPAD * 32 + 255) / 256, 256, 0, stream>>>(x, hb);
    convert_w<<<(256 * 32 + 255) / 256, 256, 0, stream>>>(Ws[0], wb, 24, 32);
    gemm_bf16<<<gemmGrid, 256, 0, stream>>>(hb, wb, tbuf, 32);
    (void)hipMemsetAsync(agg, 0, (size_t)N * HDIM * 4, stream);
    aggregate_k<<<aggBlocks, 256, 0, stream>>>(tbuf, agg, srcp, dstp, dinv, E, N);

    // ----- layers 2..6 (bias+ReLU of previous layer fused into bf16 convert)
    for (int L = 1; L < 6; ++L) {
        convert_h<<<(MPAD * HDIM + 255) / 256, 256, 0, stream>>>(agg, bs[L - 1], hb);
        convert_w<<<(256 * HDIM + 255) / 256, 256, 0, stream>>>(Ws[L], wb, HDIM, HDIM);
        gemm_bf16<<<gemmGrid, 256, 0, stream>>>(hb, wb, tbuf, HDIM);
        (void)hipMemsetAsync(agg, 0, (size_t)N * HDIM * 4, stream);
        aggregate_k<<<aggBlocks, 256, 0, stream>>>(tbuf, agg, srcp, dstp, dinv, E, N);
    }

    // ----- mean pool + head (b6 folded into head)
    (void)hipMemsetAsync(psum, 0, (size_t)G * HDIM * 4, stream);
    (void)hipMemsetAsync(cnt, 0, (size_t)G * 4, stream);
    pool_k<<<N, 256, 0, stream>>>(agg, batch, psum, cnt);
    head_k<<<G, 256, 0, stream>>>(psum, cnt, bs[5], Wlin, blin, (float*)d_out);
}
